// BondingGraphGNN_42726334660742
// MI455X (gfx1250) — matrix operand order repaired and checked by reading, not compiled, not run on hardware
//
#include <hip/hip_runtime.h>
#include <hip/hip_bf16.h>

typedef __attribute__((ext_vector_type(16))) _Float16 v16h;
typedef __attribute__((ext_vector_type(4)))  _Float16 v4h;
typedef __attribute__((ext_vector_type(8)))  float    v8f;

#define N_NODES  50000
#define N_EDGES  800000
#define FEAT     90
#define HDIM     128
#define H3       384
#define STEPS    4
#define N_GRAPHS 100

__device__ __forceinline__ int lane_id() { return (int)(threadIdx.x & 31u); }

// ---- WMMA operand loaders (CDNA5 16-bit layouts, wave32) -------------------
// A tile: 16 x 32 f16 slice at column k0 of a row-major LDS tile (row stride
// ldk). Lane L holds row M=L&15; lanes 0-15 carry K=k0+{0..7,16..23}, lanes
// 16-31 carry K=k0+{8..15,24..31}. Two 16B groups -> 2x ds_load_b128.
__device__ __forceinline__ v16h load_a(const _Float16* As, int ldk, int k0) {
  const int lane = lane_id();
  const int m  = lane & 15;
  const int kb = (lane >> 4) << 3;           // 0 or 8
  const _Float16* row = As + m * ldk + k0;
  v16h a;
#pragma unroll
  for (int e = 0; e < 8; ++e) a[e]     = row[kb + e];
#pragma unroll
  for (int e = 0; e < 8; ++e) a[8 + e] = row[16 + kb + e];
  return a;
}

// B tile from the pre-packed buffer: lane-major 16-half groups per (ct,kc)
// tile; one aligned 32B load -> two coalesced global_load_b128, no branches.
__device__ __forceinline__ v16h load_b_packed(const _Float16* __restrict__ P,
                                              int ct, int kc, int nkt) {
  const _Float16* p = P + (((size_t)(ct * nkt + kc) * 32) + lane_id()) * 16;
  return *(const v16h*)p;
}

// Pack a row-major fp32 weight W [rows x (nct*16)], row stride ld, into the
// per-lane WMMA B layout (f16, zero-padded past `rows`).
__global__ void __launch_bounds__(256)
pack_b_kernel(const float* __restrict__ W, int ld, int rows, int nkt, int total,
              _Float16* __restrict__ out) {
  const int idx = blockIdx.x * blockDim.x + threadIdx.x;
  if (idx >= total) return;
  const int e    = idx & 15;
  const int lane = (idx >> 4) & 31;
  const int tile = idx >> 9;
  const int kc   = tile % nkt;
  const int ct   = tile / nkt;
  const int n = lane & 15;
  const int k = kc * 32 + ((e & 8) << 1) + ((lane >> 4) << 3) + (e & 7);
  const float v = (k < rows) ? W[(size_t)k * ld + ct * 16 + n] : 0.f;
  out[idx] = (_Float16)v;
}

// ---------------- CSR construction (once per call, reused all 4 steps) ------
__global__ void __launch_bounds__(256)
zero_int_kernel(int* __restrict__ p, int n) {
  const int i = blockIdx.x * blockDim.x + threadIdx.x;
  if (i < n) p[i] = 0;
}

__global__ void __launch_bounds__(256)
hist_kernel(const int* __restrict__ dst, int* __restrict__ deg) {
  const int e = blockIdx.x * blockDim.x + threadIdx.x;
  if (e < N_EDGES) atomicAdd(&deg[dst[e]], 1);
}

// Single-block exclusive scan over deg -> rowstart[0..N_NODES], cursor copy.
__global__ void __launch_bounds__(256)
scan_kernel(const int* __restrict__ deg, int* __restrict__ rowstart,
            int* __restrict__ cursor) {
  __shared__ int carry;
  __shared__ int buf[256];
  if (threadIdx.x == 0) carry = 0;
  __syncthreads();
  for (int base = 0; base < N_NODES; base += 256) {
    const int i = base + (int)threadIdx.x;
    const int v = (i < N_NODES) ? deg[i] : 0;
    buf[threadIdx.x] = v;
    __syncthreads();
    for (int ofs = 1; ofs < 256; ofs <<= 1) {
      const int t = (threadIdx.x >= (unsigned)ofs) ? buf[threadIdx.x - ofs] : 0;
      __syncthreads();
      buf[threadIdx.x] += t;
      __syncthreads();
    }
    const int excl = carry + buf[threadIdx.x] - v;
    if (i < N_NODES) { rowstart[i] = excl; cursor[i] = excl; }
    __syncthreads();
    if (threadIdx.x == 255) carry += buf[255];
    __syncthreads();
  }
  if (threadIdx.x == 0) rowstart[N_NODES] = carry;   // == N_EDGES
}

__global__ void __launch_bounds__(256)
fill_kernel(const int* __restrict__ src, const int* __restrict__ dst,
            int* __restrict__ cursor, int* __restrict__ esrc) {
  const int e = blockIdx.x * blockDim.x + threadIdx.x;
  if (e >= N_EDGES) return;
  const int slot = atomicAdd(&cursor[dst[e]], 1);
  esrc[slot] = src[e];
}

// ---------------------------------------------------------------------------
// h = relu(x @ W_emb)   x:[N,90] fp32, K padded 90->96 (padding baked in Pemb)
__global__ void __launch_bounds__(128)
embed_kernel(const float* __restrict__ x, const _Float16* __restrict__ Pemb,
             float* __restrict__ h) {
  __shared__ _Float16 As[16 * 96];
  const int row0 = blockIdx.x * 16;
  for (int i = threadIdx.x; i < 16 * 96; i += blockDim.x) {
    const int r = i / 96, k = i - r * 96;
    As[i] = (k < FEAT) ? (_Float16)x[(size_t)(row0 + r) * FEAT + k] : (_Float16)0.f;
  }
  __syncthreads();
  const int wave = threadIdx.x >> 5;
  const int lane = lane_id();
  const int n  = lane & 15;
  const int mb = (lane >> 4) << 3;
#pragma unroll
  for (int t = 0; t < 2; ++t) {
    const int ct = wave * 2 + t;
    v8f acc = {};
#pragma unroll
    for (int kc = 0; kc < 3; ++kc) {
      v16h a = load_a(As, 96, kc * 32);
      v16h b = load_b_packed(Pemb, ct, kc, 3);
      acc = __builtin_amdgcn_wmma_f32_16x16x32_f16(false, a, false, b,
                                                   (short)0, acc, false, false);
    }
#pragma unroll
    for (int r = 0; r < 8; ++r) {
      const float v = acc[r];
      h[(size_t)(row0 + mb + r) * HDIM + ct * 16 + n] = v > 0.f ? v : 0.f;
    }
  }
}

// m16 = f16(h @ W_msg[s])  -- messages kept in f16: halves gather traffic,
// 12.8MB buffer stays resident in the 192MB L2 for the random row gathers.
__global__ void __launch_bounds__(128)
msg_kernel(const float* __restrict__ h, const _Float16* __restrict__ Pw,
           _Float16* __restrict__ m16) {
  __shared__ _Float16 As[16 * HDIM];
  const int row0 = blockIdx.x * 16;
  for (int i = threadIdx.x; i < 16 * HDIM; i += blockDim.x) {
    const int r = i >> 7, k = i & 127;
    As[i] = (_Float16)h[(size_t)(row0 + r) * HDIM + k];
  }
  __syncthreads();
  const int wave = threadIdx.x >> 5;
  const int lane = lane_id();
  const int n  = lane & 15;
  const int mb = (lane >> 4) << 3;
#pragma unroll
  for (int t = 0; t < 2; ++t) {
    const int ct = wave * 2 + t;
    v8f acc = {};
#pragma unroll
    for (int kc = 0; kc < 4; ++kc) {
      v16h a = load_a(As, HDIM, kc * 32);
      v16h b = load_b_packed(Pw, ct, kc, 4);
      acc = __builtin_amdgcn_wmma_f32_16x16x32_f16(false, a, false, b,
                                                   (short)0, acc, false, false);
    }
#pragma unroll
    for (int r = 0; r < 8; ++r)
      m16[(size_t)(row0 + mb + r) * HDIM + ct * 16 + n] = (_Float16)acc[r];
  }
}

// agg[n] = sum over in-edges of m16[src]. One wave per node, lane owns 4
// features: per edge a wave reads one contiguous 256B f16 row (b64/lane),
// accumulates f32 in registers, then one coalesced float4 store. No atomics.
__global__ void __launch_bounds__(256)
gather_kernel(const _Float16* __restrict__ m16, const int* __restrict__ rowstart,
              const int* __restrict__ esrc, float* __restrict__ agg) {
  const int wave = threadIdx.x >> 5;
  const int lane = lane_id();
  const int node = blockIdx.x * 8 + wave;    // 6250 * 8 = 50000 exact
  const int beg = rowstart[node];
  const int end = rowstart[node + 1];
  float ax = 0.f, ay = 0.f, az = 0.f, aw = 0.f;
  for (int e = beg; e < end; ++e) {
    const int s = esrc[e];
    const v4h v = *(const v4h*)(m16 + (size_t)s * HDIM + lane * 4);
    ax += (float)v[0]; ay += (float)v[1]; az += (float)v[2]; aw += (float)v[3];
  }
  float4 o; o.x = ax; o.y = ay; o.z = az; o.w = aw;
  *(float4*)(agg + (size_t)node * HDIM + lane * 4) = o;
}

// Fused GRU: gx = agg@W_ih + b_ih, gh = h@W_hh + b_hh (WMMA, packed weights),
// then gates, in-place h update. One 16-row tile per block, 4 waves x 12 tiles.
__global__ void __launch_bounds__(128)
gru_kernel(const float* __restrict__ agg, float* __restrict__ h,
           const _Float16* __restrict__ Pih, const _Float16* __restrict__ Phh,
           const float* __restrict__ b_ih, const float* __restrict__ b_hh) {
  __shared__ _Float16 Aagg[16 * HDIM];
  __shared__ _Float16 Ah[16 * HDIM];
  __shared__ float G[16 * 768];              // [r][0:384]=gx  [r][384:768]=gh
  const int row0 = blockIdx.x * 16;
  for (int i = threadIdx.x; i < 16 * HDIM; i += blockDim.x) {
    const int r = i >> 7, k = i & 127;
    Aagg[i] = (_Float16)agg[(size_t)(row0 + r) * HDIM + k];
    Ah[i]   = (_Float16)h[(size_t)(row0 + r) * HDIM + k];
  }
  __syncthreads();
  const int wave = threadIdx.x >> 5;
  const int lane = lane_id();
  const int n  = lane & 15;
  const int mb = (lane >> 4) << 3;
  for (int p = wave; p < 48; p += 4) {       // 48 = 2 matrices x 24 col tiles
    const int mat = p / 24;                  // 0: input path, 1: hidden path
    const int ct  = p % 24;
    const _Float16* Pw  = mat ? Phh : Pih;
    const float*   bias = mat ? b_hh : b_ih;
    const _Float16* A   = mat ? Ah : Aagg;
    v8f acc = {};
#pragma unroll
    for (int kc = 0; kc < 4; ++kc) {
      v16h a = load_a(A, HDIM, kc * 32);
      v16h b = load_b_packed(Pw, ct, kc, 4);
      acc = __builtin_amdgcn_wmma_f32_16x16x32_f16(false, a, false, b,
                                                   (short)0, acc, false, false);
    }
    const float bv = bias[ct * 16 + n];
#pragma unroll
    for (int r = 0; r < 8; ++r)
      G[(mb + r) * 768 + mat * H3 + ct * 16 + n] = acc[r] + bv;
  }
  __syncthreads();
  for (int i = threadIdx.x; i < 16 * HDIM; i += blockDim.x) {
    const int r = i >> 7, f = i & 127;
    const float* g = G + r * 768;
    const float rr = 1.f / (1.f + __expf(-(g[f]            + g[H3 + f])));
    const float zz = 1.f / (1.f + __expf(-(g[HDIM + f]     + g[H3 + HDIM + f])));
    const float nn = tanhf(g[2 * HDIM + f] + rr * g[H3 + 2 * HDIM + f]);
    const size_t idx = (size_t)(row0 + r) * HDIM + f;
    const float hv = h[idx];
    h[idx] = (1.f - zz) * nn + zz * hv;
  }
}

__global__ void __launch_bounds__(256)
zero_kernel(float* __restrict__ p, int n) {
  const int i = blockIdx.x * blockDim.x + threadIdx.x;
  if (i < n) p[i] = 0.f;
}

// pooled[batch[n]] += relu(h[n]); counts[batch[n]] += 1
__global__ void __launch_bounds__(256)
pool_kernel(const float* __restrict__ h, const int* __restrict__ batch,
            float* __restrict__ pooled, float* __restrict__ counts) {
  const int tid = blockIdx.x * blockDim.x + threadIdx.x;
  if (tid >= N_NODES * HDIM) return;
  const int node = tid >> 7, f = tid & 127;
  const int g = batch[node];
  float v = h[tid];
  v = v > 0.f ? v : 0.f;
  atomicAdd(pooled + g * HDIM + f, v);
  if (f == 0) atomicAdd(counts + g, 1.f);
}

// out[g] = softplus(relu(mean_pooled @ W1 + b1) @ W2 + b2)
__global__ void __launch_bounds__(128)
mlp_kernel(const float* __restrict__ pooled, const float* __restrict__ counts,
           const float* __restrict__ W1, const float* __restrict__ b1,
           const float* __restrict__ W2, const float* __restrict__ b2,
           float* __restrict__ out) {
  __shared__ float p[HDIM];
  __shared__ float red[HDIM];
  const int g = blockIdx.x;
  const int t = threadIdx.x;
  float cnt = counts[g];
  cnt = cnt > 1.f ? cnt : 1.f;
  p[t] = pooled[g * HDIM + t] / cnt;
  __syncthreads();
  float acc = b1[t];
#pragma unroll 8
  for (int k = 0; k < HDIM; ++k) acc += p[k] * W1[k * HDIM + t];
  acc = acc > 0.f ? acc : 0.f;
  red[t] = acc * W2[t];
  __syncthreads();
  for (int s = 64; s > 0; s >>= 1) {
    if (t < s) red[t] += red[t + s];
    __syncthreads();
  }
  if (t == 0) {
    const float z = red[0] + b2[0];
    out[g] = log1pf(__expf(z));
  }
}

extern "C" void kernel_launch(void* const* d_in, const int* in_sizes, int n_in,
                              void* d_out, int out_size, void* d_ws, size_t ws_size,
                              hipStream_t stream) {
  const float* x     = (const float*)d_in[0];
  const int*   eidx  = (const int*)d_in[1];
  const int*   batch = (const int*)d_in[2];
  const float* W_emb = (const float*)d_in[3];
  const float* W_msg = (const float*)d_in[4];
  const float* W_ih  = (const float*)d_in[5];
  const float* W_hh  = (const float*)d_in[6];
  const float* b_ih  = (const float*)d_in[7];
  const float* b_hh  = (const float*)d_in[8];
  const float* W1    = (const float*)d_in[9];
  const float* b1    = (const float*)d_in[10];
  const float* W2    = (const float*)d_in[11];
  const float* b2    = (const float*)d_in[12];
  float* out = (float*)d_out;

  char* ws = (char*)d_ws;
  const size_t nodeBytes = (size_t)N_NODES * HDIM * sizeof(float);   // 25.6 MB
  size_t off = 0;
  float*    h    = (float*)(ws + off);    off += nodeBytes;
  float*    agg  = (float*)(ws + off);    off += nodeBytes;
  _Float16* m16  = (_Float16*)(ws + off); off += (size_t)N_NODES * HDIM * 2;
  float*    pooled = (float*)(ws + off);  off += (size_t)N_GRAPHS * HDIM * 4;
  float*    counts = (float*)(ws + off);  off += (size_t)N_GRAPHS * 4;
  int* deg      = (int*)(ws + off);       off += (size_t)N_NODES * 4;
  int* cursor   = (int*)(ws + off);       off += (size_t)N_NODES * 4;
  int* rowstart = (int*)(ws + off);       off += (size_t)(N_NODES + 1) * 4;
  int* esrc     = (int*)(ws + off);       off += (size_t)N_EDGES * 4;
  off = (off + 511) & ~(size_t)511;       // 32B+ alignment for v16h loads
  const int embTotal = 3 * 8 * 512;
  const int msgTotal = 4 * 8 * 512;
  const int gruTotal = 4 * 24 * 512;
  _Float16* Pemb = (_Float16*)(ws + off); off += (size_t)embTotal * 2;
  _Float16* Pmsg = (_Float16*)(ws + off); off += (size_t)STEPS * msgTotal * 2;
  _Float16* Pih  = (_Float16*)(ws + off); off += (size_t)gruTotal * 2;
  _Float16* Phh  = (_Float16*)(ws + off); off += (size_t)gruTotal * 2;

  const int* src = eidx;
  const int* dst = eidx + N_EDGES;
  const int rowTiles = N_NODES / 16;      // 3125 exact
  const int edgeBlocks = (N_EDGES + 255) / 256;

  // ---- pack weights into WMMA B-operand layout ----
  pack_b_kernel<<<(embTotal + 255) / 256, 256, 0, stream>>>(W_emb, HDIM, FEAT, 3,
                                                            embTotal, Pemb);
  for (int s = 0; s < STEPS; ++s)
    pack_b_kernel<<<(msgTotal + 255) / 256, 256, 0, stream>>>(
        W_msg + (size_t)s * HDIM * HDIM, HDIM, HDIM, 4, msgTotal,
        Pmsg + (size_t)s * msgTotal);
  pack_b_kernel<<<(gruTotal + 255) / 256, 256, 0, stream>>>(W_ih, H3, HDIM, 4,
                                                            gruTotal, Pih);
  pack_b_kernel<<<(gruTotal + 255) / 256, 256, 0, stream>>>(W_hh, H3, HDIM, 4,
                                                            gruTotal, Phh);

  // ---- CSR build (once per call; reused for all 4 steps) ----
  zero_int_kernel<<<(N_NODES + 255) / 256, 256, 0, stream>>>(deg, N_NODES);
  hist_kernel<<<edgeBlocks, 256, 0, stream>>>(dst, deg);
  scan_kernel<<<1, 256, 0, stream>>>(deg, rowstart, cursor);
  fill_kernel<<<edgeBlocks, 256, 0, stream>>>(src, dst, cursor, esrc);

  // ---- pipeline ----
  embed_kernel<<<rowTiles, 128, 0, stream>>>(x, Pemb, h);
  for (int s = 0; s < STEPS; ++s) {
    msg_kernel<<<rowTiles, 128, 0, stream>>>(h, Pmsg + (size_t)s * msgTotal, m16);
    gather_kernel<<<N_NODES / 8, 256, 0, stream>>>(m16, rowstart, esrc, agg);
    gru_kernel<<<rowTiles, 128, 0, stream>>>(agg, h, Pih, Phh, b_ih, b_hh);
  }

  zero_kernel<<<(N_GRAPHS * HDIM + N_GRAPHS + 255) / 256, 256, 0, stream>>>(
      pooled, N_GRAPHS * HDIM + N_GRAPHS);
  pool_kernel<<<(N_NODES * HDIM + 255) / 256, 256, 0, stream>>>(h, batch, pooled, counts);
  mlp_kernel<<<N_GRAPHS, 128, 0, stream>>>(pooled, counts, W1, b1, W2, b2, out);
}
